// BahdanauAttention_77077483094475
// MI455X (gfx1250) — matrix operand order repaired and checked by reading, not compiled, not run on hardware
//
#include <hip/hip_runtime.h>
#include <hip/hip_bf16.h>
#include <math.h>

// ---------------------------------------------------------------------------
// Bahdanau attention w/ coverage, CDNA5 (gfx1250), wave32 + bf16 WMMA.
// B=64, S=2048, E=1024, D=512, A=1024.
// ---------------------------------------------------------------------------

#define B_DIM   64
#define S_LEN   2048
#define E_DIM   1024
#define D_DIM   512
#define A_DIM   1024

typedef __bf16 bf16_t;
typedef bf16_t v16bf __attribute__((ext_vector_type(16)));
typedef bf16_t v8bf  __attribute__((ext_vector_type(8)));
typedef float  v8f   __attribute__((ext_vector_type(8)));

__device__ __forceinline__ unsigned short f2bf(float f) {
  unsigned u = __float_as_uint(f);
  unsigned r = u + 0x7FFFu + ((u >> 16) & 1u);   // round-to-nearest-even
  return (unsigned short)(r >> 16);
}

// Hardware V_TANH_F32 on gfx1250 when available; avoids libm's branchy
// exp-based expansion (divergent EXEC churn between WMMA bursts).
__device__ __forceinline__ float hw_tanh(float x) {
#if defined(__has_builtin)
#if __has_builtin(__builtin_amdgcn_tanhf)
  return __builtin_amdgcn_tanhf(x);
#else
  return tanhf(x);
#endif
#else
  return tanhf(x);
#endif
}

// --------------------------- kernel 0: W_encoder fp32 -> bf16 ---------------
__global__ __launch_bounds__(256) void cvt_w_kernel(const float* __restrict__ W,
                                                    unsigned short* __restrict__ Wb,
                                                    int n) {
  int i = blockIdx.x * 256 + threadIdx.x;
  if (i < n) Wb[i] = f2bf(W[i]);
}

// --------------------------- kernel 1: dec_feat = dh @ Wd^T -----------------
__global__ __launch_bounds__(128) void decfeat_kernel(const float* __restrict__ dh,
                                                      const float* __restrict__ Wd,
                                                      float* __restrict__ dec_feat) {
  __shared__ float h_sh[D_DIM];
  const int b = blockIdx.y;
  const int a = blockIdx.x * 128 + threadIdx.x;
  for (int e = threadIdx.x; e < D_DIM; e += 128) h_sh[e] = dh[b * D_DIM + e];
  __syncthreads();
  const float* w = Wd + (size_t)a * D_DIM;
  float acc = 0.f;
#pragma unroll 4
  for (int e = 0; e < D_DIM; ++e) acc += h_sh[e] * w[e];
  dec_feat[b * A_DIM + a] = acc;
}

// --------------------------- kernel 2: fused scores (WMMA) ------------------
// One block = 4 waves = one 16-row M-tile of (b,s). Each wave handles 16 of the
// 64 N-tiles; C-tile epilogue: tanh(enc_feat + dec_feat + cov*wcov) dot v.
__global__ __launch_bounds__(128) void scores_kernel(
    const float* __restrict__ enc,            // [B, S, E]
    const float* __restrict__ coverage,       // [B, S]
    const unsigned short* __restrict__ Wb,    // [A, E] bf16 bits
    const float* __restrict__ dec_feat,       // [B, A]
    const float* __restrict__ Wcov,           // [A]
    const float* __restrict__ vvec,           // [A]
    float* __restrict__ scores)               // [B, S]
{
  __shared__ __align__(16) unsigned short Ald[16 * E_DIM];  // 32 KB bf16 A-tile
  __shared__ float wsum[4][16];

  const int tid  = threadIdx.x;
  const int wave = tid >> 5;
  const int lane = tid & 31;

  const int mtile = blockIdx.x;              // 0..8191
  const int b  = mtile / (S_LEN / 16);
  const int s0 = (mtile % (S_LEN / 16)) * 16;

  // Stage 16x1024 fp32 rows -> bf16 LDS (row-major, stride E_DIM).
  const float* src = enc + ((size_t)b * S_LEN + s0) * E_DIM;
  for (int i = tid; i < 16 * E_DIM / 4; i += 128) {
    float4 f = ((const float4*)src)[i];
    ushort4 h;
    h.x = f2bf(f.x); h.y = f2bf(f.y); h.z = f2bf(f.z); h.w = f2bf(f.w);
    ((ushort4*)Ald)[i] = h;
  }
  __syncthreads();

  const int half = lane >> 4;                // 0: lanes 0-15, 1: lanes 16-31
  const int lmod = lane & 15;
  const int arow = lmod;                     // A-fragment row (M = lane%16)
  const int ac   = half * 16;                // A-fragment K sub-offset

  // coverage per local row (uniform across each 16-lane half)
  float cov[8];
#pragma unroll
  for (int r = 0; r < 8; ++r)
    cov[r] = coverage[(size_t)b * S_LEN + s0 + r + half * 8];

  float sacc[8];
#pragma unroll
  for (int r = 0; r < 8; ++r) sacc[r] = 0.f;

  for (int nt = wave * 16; nt < wave * 16 + 16; ++nt) {
    const int n0   = nt * 16;
    const int ncol = n0 + lmod;              // B-fragment column (N = lane%16)
    const unsigned short* brow = Wb + (size_t)ncol * E_DIM + half * 16;

    v8f c = {};
#pragma unroll 4
    for (int kt = 0; kt < E_DIM / 32; ++kt) {
      const int k0 = kt * 32;
      union { v16bf v; v8bf p[2]; } a;
      // lane<16: K = k0+0..15 ; lane>=16: K = k0+16..31
      a.p[0] = *(const v8bf*)&Ald[arow * E_DIM + k0 + ac];
      a.p[1] = *(const v8bf*)&Ald[arow * E_DIM + k0 + ac + 8];
      // B column ncol, K = k0 + 16*half .. +16 (32 contiguous bytes)
      v16bf bb = *(const v16bf*)(brow + k0);
      c = __builtin_amdgcn_wmma_f32_16x16x32_bf16(false, a.v, false, bb,
                                                  (short)0, c, false, false);
    }

    const float dv = dec_feat[b * A_DIM + ncol];
    const float wc = Wcov[ncol];
    const float vv = vvec[ncol];
#pragma unroll
    for (int r = 0; r < 8; ++r) {
      // C layout: VGPR r, lanes 0-15 -> M=r, lanes 16-31 -> M=r+8; N=lane%16
      float x = c[r] + dv + cov[r] * wc;
      sacc[r] += vv * hw_tanh(x);
    }
  }

  // Reduce over the 16 N-lanes of each half.
#pragma unroll
  for (int r = 0; r < 8; ++r) {
    float s = sacc[r];
    s += __shfl_xor(s, 1, 16);
    s += __shfl_xor(s, 2, 16);
    s += __shfl_xor(s, 4, 16);
    s += __shfl_xor(s, 8, 16);
    sacc[r] = s;
  }
  if (lmod == 0) {                            // lanes 0 and 16
#pragma unroll
    for (int r = 0; r < 8; ++r) wsum[wave][half * 8 + r] = sacc[r];
  }
  __syncthreads();

  if (tid < 16) {
    float s = wsum[0][tid] + wsum[1][tid] + wsum[2][tid] + wsum[3][tid];
    scores[(size_t)b * S_LEN + s0 + tid] = s;
  }
}

// --------------------------- kernel 3: masked softmax over S ----------------
__global__ __launch_bounds__(256) void softmax_kernel(const float* __restrict__ scores,
                                                      const int* __restrict__ mask,
                                                      float* __restrict__ attn) {
  __shared__ float sh[S_LEN];
  __shared__ float red[8];
  const int b = blockIdx.x, tid = threadIdx.x;

  float lmax = -INFINITY;
  for (int s = tid; s < S_LEN; s += 256) {
    float sc = scores[(size_t)b * S_LEN + s];
    if (mask[(size_t)b * S_LEN + s] == 0) sc = -1e10f;
    sh[s] = sc;
    lmax = fmaxf(lmax, sc);
  }
  for (int off = 16; off >= 1; off >>= 1) lmax = fmaxf(lmax, __shfl_xor(lmax, off, 32));
  if ((tid & 31) == 0) red[tid >> 5] = lmax;
  __syncthreads();
  float bmax = red[0];
#pragma unroll
  for (int i = 1; i < 8; ++i) bmax = fmaxf(bmax, red[i]);
  __syncthreads();

  float lsum = 0.f;
  for (int s = tid; s < S_LEN; s += 256) {
    float e = __expf(sh[s] - bmax);
    sh[s] = e;
    lsum += e;
  }
  for (int off = 16; off >= 1; off >>= 1) lsum += __shfl_xor(lsum, off, 32);
  if ((tid & 31) == 0) red[tid >> 5] = lsum;
  __syncthreads();
  float bsum = 0.f;
#pragma unroll
  for (int i = 0; i < 8; ++i) bsum += red[i];
  const float inv = 1.0f / bsum;
  for (int s = tid; s < S_LEN; s += 256)
    attn[(size_t)b * S_LEN + s] = sh[s] * inv;
}

// --------------------------- kernel 4: context = attn @ enc -----------------
__global__ __launch_bounds__(256) void context_kernel(const float* __restrict__ enc,
                                                      const float* __restrict__ attn,
                                                      float* __restrict__ ctx) {
  __shared__ float a_sh[S_LEN];
  const int b = blockIdx.x;
  const int e = blockIdx.y * 256 + threadIdx.x;
  for (int s = threadIdx.x; s < S_LEN; s += 256)
    a_sh[s] = attn[(size_t)b * S_LEN + s];
  __syncthreads();
  const float* ep = enc + (size_t)b * S_LEN * E_DIM + e;
  float acc = 0.f;
#pragma unroll 4
  for (int s = 0; s < S_LEN; ++s) acc += a_sh[s] * ep[(size_t)s * E_DIM];
  ctx[b * E_DIM + e] = acc;
}

// ---------------------------------------------------------------------------
extern "C" void kernel_launch(void* const* d_in, const int* in_sizes, int n_in,
                              void* d_out, int out_size, void* d_ws, size_t ws_size,
                              hipStream_t stream) {
  const float* decoder_hidden  = (const float*)d_in[0];   // [B, D]
  const float* encoder_outputs = (const float*)d_in[1];   // [B, S, E]
  const int*   encoder_mask    = (const int*)d_in[2];     // [B, S]
  const float* coverage        = (const float*)d_in[3];   // [B, S]
  const float* W_encoder       = (const float*)d_in[4];   // [A, E]
  const float* W_decoder       = (const float*)d_in[5];   // [A, D]
  const float* W_coverage      = (const float*)d_in[6];   // [A, 1]
  const float* v               = (const float*)d_in[7];   // [1, A]

  float* ctx  = (float*)d_out;                            // [B, E]
  float* attn = (float*)d_out + (size_t)B_DIM * E_DIM;    // [B, S]

  // workspace layout
  unsigned short* Wb = (unsigned short*)d_ws;                               // 2 MB
  float* dec_feat = (float*)((char*)d_ws + (size_t)A_DIM * E_DIM * 2);     // 256 KB
  float* scores   = dec_feat + (size_t)B_DIM * A_DIM;                      // 512 KB

  // 0) W_encoder -> bf16
  cvt_w_kernel<<<(A_DIM * E_DIM + 255) / 256, 256, 0, stream>>>(
      W_encoder, Wb, A_DIM * E_DIM);

  // 1) dec_feat
  decfeat_kernel<<<dim3(A_DIM / 128, B_DIM), 128, 0, stream>>>(
      decoder_hidden, W_decoder, dec_feat);

  // 2) fused scores via bf16 WMMA
  scores_kernel<<<(B_DIM * S_LEN) / 16, 128, 0, stream>>>(
      encoder_outputs, coverage, Wb, dec_feat, W_coverage, v, scores);

  // 3) masked softmax -> attn
  softmax_kernel<<<B_DIM, 256, 0, stream>>>(scores, encoder_mask, attn);

  // 4) context
  context_kernel<<<dim3(B_DIM, E_DIM / 256), 256, 0, stream>>>(
      encoder_outputs, attn, ctx);
}
